// PointPillars_32933809226148
// MI455X (gfx1250) — compile-verified, matrix operand order checked by reading, not run on hardware
//
#include <hip/hip_runtime.h>

typedef __attribute__((ext_vector_type(16))) _Float16 v16h;
typedef __attribute__((ext_vector_type(8)))  float    v8f;

#define BATCH   8
#define NPILLAR 12000
#define MPTS    32
#define FDIM    9
#define CCH     64
#define XN      144
#define YN      144
#define HW      (XN * YN)
#define PILLARS_PER_WAVE 8

// Branchless 16x32 f16 A-fragment build (K padded 9->32 with zeros).
// VGPR layout (CDNA5 ISA, 16-bit A 16x32):
//   lanes 0-15 : M = lane,    halves 0..7 = K0..7,  halves 8..15 = K16..23 (zero)
//   lanes 16-31: M = lane-16, halves 0..7 = K8..15 (only K8 valid), rest zero
// All lanes issue the same two b128 loads; hi lanes read floats 5..8 (in
// bounds: each row has 9 floats) and select f8 into half 0.
__device__ inline v16h load_a_tile(const float* __restrict__ pillars,
                                   int rowbase, int n, bool lo) {
    const float* rp = pillars + (size_t)(rowbase + n) * FDIM + (lo ? 0 : 5);
    float4 v0 = *(const float4*)rp;        // lo: f0..f3   hi: f5..f8
    float4 v1 = *(const float4*)(rp + (lo ? 4 : 0));  // lo: f4..f7  hi: dup

    v16h a = {};
    a[0] = (_Float16)(lo ? v0.x : v0.w);   // lo: f0       hi: f8
    a[1] = (_Float16)(lo ? v0.y : 0.0f);
    a[2] = (_Float16)(lo ? v0.z : 0.0f);
    a[3] = (_Float16)(lo ? v0.w : 0.0f);
    a[4] = (_Float16)(lo ? v1.x : 0.0f);
    a[5] = (_Float16)(lo ? v1.y : 0.0f);
    a[6] = (_Float16)(lo ? v1.z : 0.0f);
    a[7] = (_Float16)(lo ? v1.w : 0.0f);
    return a;
}

__global__ __launch_bounds__(256)
void pp_fused_wmma_kernel(const float* __restrict__ pillars,
                          const int*   __restrict__ pidx,     // (B,P,3)
                          const float* __restrict__ conv_w,   // (64,9)
                          const float* __restrict__ bn_gamma,
                          const float* __restrict__ bn_beta,
                          const float* __restrict__ bn_mean,
                          const float* __restrict__ bn_var,
                          float* __restrict__ out)            // (B,64,144,144)
{
    const int lane  = threadIdx.x & 31;
    const int wv    = blockIdx.x * 8 + (threadIdx.x >> 5);   // 12000 waves
    const int p0    = wv * PILLARS_PER_WAVE;                 // first pillar
    const int n     = lane & 15;
    const bool lo   = lane < 16;

    // NPILLAR % PILLARS_PER_WAVE == 0 -> a wave never crosses a batch boundary
    const int b = p0 / NPILLAR;
    float* const obb = out + (size_t)b * CCH * HW;

    // ---- Hoisted: B fragments (conv_w^T), four 32x16 f16 tiles ----
    // Mirror of the A layout: column N = lane&15; lanes 0-15 hold K0..7,
    // lanes 16-31 hold K8..15 (only K8 valid); K16..31 zero-padded.
    v16h bfrag[4];
#pragma unroll
    for (int nt = 0; nt < 4; ++nt) {
        v16h bf = {};
        const float* wp = conv_w + (size_t)(nt * 16 + n) * FDIM;
        if (lo) {
#pragma unroll
            for (int k = 0; k < 8; ++k) bf[k] = (_Float16)wp[k];
        } else {
            bf[0] = (_Float16)wp[8];
        }
        bfrag[nt] = bf;
    }

    // ---- Hoisted: BN folded to y = m*scale + shift per channel ----
    float bscale[4], bshift[4];
#pragma unroll
    for (int nt = 0; nt < 4; ++nt) {
        const int c = nt * 16 + n;
        const float inv_std = rsqrtf(bn_var[c] + 1e-5f);
        bscale[nt] = bn_gamma[c] * inv_std;
        bshift[nt] = bn_beta[c] - bn_mean[c] * bscale[nt];
    }

    // ---- Per-pillar loop: contiguous 8-pillar chunk per wave ----
    for (int pp = 0; pp < PILLARS_PER_WAVE; ++pp) {
        const int pillar  = p0 + pp;
        const int rowbase = pillar * MPTS;

        v16h a0 = load_a_tile(pillars, rowbase,      n, lo);
        v16h a1 = load_a_tile(pillars, rowbase + 16, n, lo);

        const int yi = pidx[pillar * 3 + 1];
        int       xi = pidx[pillar * 3 + 2];
        xi = xi < 0 ? 0 : (xi > XN - 1 ? XN - 1 : xi);
        const int flat = yi * XN + xi;

        // Issue all 8 WMMAs back-to-back (independent accumulators) so the
        // WMMA->VALU hazard slots are filled by other WMMAs, not v_nops.
        v8f acc0[4], acc1[4];
#pragma unroll
        for (int nt = 0; nt < 4; ++nt) {
            v8f z0 = {};
            v8f z1 = {};
            acc0[nt] = __builtin_amdgcn_wmma_f32_16x16x32_f16(
                           false, a0, false, bfrag[nt], (short)0, z0, false, false);
            acc1[nt] = __builtin_amdgcn_wmma_f32_16x16x32_f16(
                           false, a1, false, bfrag[nt], (short)0, z1, false, false);
        }

        float yv[4];
#pragma unroll
        for (int nt = 0; nt < 4; ++nt) {
            // max over the 16 C-rows this lane holds (8 per tile)
            float m = fmaxf(acc0[nt][0], acc1[nt][0]);
#pragma unroll
            for (int r = 1; r < 8; ++r)
                m = fmaxf(m, fmaxf(acc0[nt][r], acc1[nt][r]));
            // combine the two lane halves (rows 0-7 vs 8-15 of each tile)
            m = fmaxf(m, __shfl_xor(m, 16, 32));
            // BN + ReLU (commute with max: monotone affine, scale > 0)
            yv[nt] = fmaxf(m * bscale[nt] + bshift[nt], 0.0f);
        }

        // single guarded store region: 4 stores, one exec toggle
        if (lo) {
            float* op = obb + (size_t)n * HW + flat;
#pragma unroll
            for (int nt = 0; nt < 4; ++nt)
                op[(size_t)(nt * 16) * HW] = yv[nt];
        }
    }
}

__global__ __launch_bounds__(256)
void zero_canvas_kernel(float4* __restrict__ p, int n4) {
    int i = blockIdx.x * blockDim.x + threadIdx.x;
    if (i < n4) p[i] = make_float4(0.f, 0.f, 0.f, 0.f);
}

extern "C" void kernel_launch(void* const* d_in, const int* in_sizes, int n_in,
                              void* d_out, int out_size, void* d_ws, size_t ws_size,
                              hipStream_t stream) {
    const float* pillars  = (const float*)d_in[0];
    const int*   pidx     = (const int*)  d_in[1];
    const float* conv_w   = (const float*)d_in[2];
    const float* bn_gamma = (const float*)d_in[3];
    const float* bn_beta  = (const float*)d_in[4];
    const float* bn_mean  = (const float*)d_in[5];
    const float* bn_var   = (const float*)d_in[6];
    float*       out      = (float*)d_out;

    // zero the 8x64x144x144 canvas (out_size = 10,616,832, divisible by 4)
    const int n4 = out_size / 4;
    zero_canvas_kernel<<<(n4 + 255) / 256, 256, 0, stream>>>((float4*)d_out, n4);

    // 96000 pillars / 8 per wave = 12000 waves; 8 waves/block -> 1500 blocks
    const int total_waves = (BATCH * NPILLAR) / PILLARS_PER_WAVE;
    pp_fused_wmma_kernel<<<total_waves / 8, 256, 0, stream>>>(
        pillars, pidx, conv_w, bn_gamma, bn_beta, bn_mean, bn_var, out);
}
